// seqKAN_wide_59794534695204
// MI455X (gfx1250) — compile-verified, matrix operand order checked by reading, not run on hardware
//
#include <hip/hip_runtime.h>

typedef __attribute__((ext_vector_type(16))) __bf16 v16bf;
typedef __attribute__((ext_vector_type(8)))  __bf16 v8bf;
typedef __attribute__((ext_vector_type(8)))  float  v8f;

#define NIN   32
#define NHID  128
#define NOUT  16
#define BATCH 256
#define SEQL  512
#define NB    9            // features per scalar input: silu + 8 spline bases
#define D1    (NIN + NHID) // 160
#define K1    (D1 * NB)    // 1440
#define KT1   (K1 / 32)    // 45 K-tiles
#define D2    NHID
#define K2    (D2 * NB)    // 1152
#define KT2   (K2 / 32)    // 36 K-tiles
#define NT1   (NHID / 16)  // 8 N-tiles == 8 waves
#define ASTRIDE 1448       // padded LDS row stride (elems), 16B-aligned rows

#define W1P_ELEMS (NT1 * KT1 * 32 * 16)   // 184320 bf16
#define W2P_ELEMS (KT2 * 32 * 16)         // 18432 bf16
#define W1P_BYTES (W1P_ELEMS * 2)         // 368640 B

// ---------- spline helpers ----------
// Uniform extended knots: t[m] = -8.8 + 1.6*m, m = 0..11 (K=3, G=5, range [-4,4])
__device__ __forceinline__ void bspline8(float x, float* B8) {
  float B[11];
#pragma unroll
  for (int m = 0; m < 11; ++m) {
    float tm = -8.8f + 1.6f * m;
    B[m] = (x >= tm && x < tm + 1.6f) ? 1.0f : 0.0f;
  }
#pragma unroll
  for (int d = 1; d <= 3; ++d) {
    float inv = 1.0f / (1.6f * (float)d);
#pragma unroll
    for (int m = 0; m < 11 - d; ++m) {
      float tm = -8.8f + 1.6f * m;
      B[m] = (x - tm) * inv * B[m] + ((tm + 1.6f * (d + 1)) - x) * inv * B[m + 1];
    }
  }
#pragma unroll
  for (int m = 0; m < 8; ++m) B8[m] = B[m];
}

__device__ __forceinline__ float silu_f(float v) {
  return v / (1.0f + __expf(-v));
}

// ---------- weight pack kernels: fuse (sb, coef*sp) and swizzle into WMMA-B order ----------
// Packed layout: Wp[nt][kt][lane][e], each lane's 16 bf16 contiguous (32B coalesced).
// lane<16: col n=nt*16+lane,    e<8 -> K=kt*32+e,    e>=8 -> K=kt*32+8+e  (i.e. 16..23)
// lane>=16: col n=nt*16+lane-16, e<8 -> K=kt*32+8+e(=8..15), e>=8 -> K=kt*32+16+e(=24..31)
__global__ void pack_w1(const float* __restrict__ coef, const float* __restrict__ sb,
                        const float* __restrict__ sp, __bf16* __restrict__ Wp) {
  int idx = blockIdx.x * 256 + threadIdx.x;
  if (idx >= W1P_ELEMS) return;
  int e    = idx & 15;
  int lane = (idx >> 4) & 31;
  int kt   = (idx >> 9) % KT1;
  int nt   = idx / (KT1 * 512);
  int n    = nt * 16 + (lane & 15);
  int half = lane >> 4;
  int kk   = kt * 32 + half * 8 + ((e < 8) ? e : (8 + e));
  int i = kk / NB, k = kk % NB;
  float v = (k == 0) ? sb[i * NHID + n]
                     : coef[(i * NHID + n) * 8 + (k - 1)] * sp[i * NHID + n];
  Wp[idx] = (__bf16)v;
}

__global__ void pack_w2(const float* __restrict__ coef, const float* __restrict__ sb,
                        const float* __restrict__ sp, __bf16* __restrict__ Wp) {
  int idx = blockIdx.x * 256 + threadIdx.x;
  if (idx >= W2P_ELEMS) return;
  int e    = idx & 15;
  int lane = (idx >> 4) & 31;
  int kt   = idx >> 9;
  int n    = lane & 15;
  int half = lane >> 4;
  int kk   = kt * 32 + half * 8 + ((e < 8) ? e : (8 + e));
  int i = kk / NB, k = kk % NB;
  float v = (k == 0) ? sb[i * NOUT + n]
                     : coef[(i * NOUT + n) * 8 + (k - 1)] * sp[i * NOUT + n];
  Wp[idx] = (__bf16)v;
}

// ---------- A-fragment load from LDS (two 16B-aligned ds_read_b128) ----------
__device__ __forceinline__ v16bf loadA(const __bf16* Ab, int rowoff, int koffA, int kt) {
  const __bf16* p = Ab + rowoff + kt * 32 + koffA;
  v8bf lo = *(const v8bf*)p;          // K kt*32 + koffA + 0..7
  v8bf hi = *(const v8bf*)(p + 16);   // K kt*32 + koffA + 16..23
  return __builtin_shufflevector(lo, hi, 0, 1, 2, 3, 4, 5, 6, 7,
                                 8, 9, 10, 11, 12, 13, 14, 15);
}

// ---------- persistent recurrence kernel: one WG per 16 batch rows ----------
__global__ void __launch_bounds__(256)
kan_rnn_kernel(const float* __restrict__ x, const __bf16* __restrict__ W1p,
               const __bf16* __restrict__ W2p, float* __restrict__ out) {
  __shared__ __attribute__((aligned(16))) __bf16 Abuf[16 * ASTRIDE]; // feature tile (bf16)
  __shared__ float hbuf[16 * NHID];                                  // hidden state (f32)

  const int tid  = threadIdx.x;
  const int lane = tid & 31;
  const int wv   = tid >> 5;          // wave id = N-tile id (0..7)
  const int b0   = blockIdx.x * 16;   // first batch row of this tile

  for (int i = tid; i < 16 * NHID; i += 256) hbuf[i] = 0.0f; // h0 = 0
  __syncthreads();

  const int half   = lane >> 4;
  const int rowoff = (lane & 15) * ASTRIDE;
  const int koffA  = half * 8;
  const __bf16* Wn = W1p + (size_t)wv * (KT1 * 512);

  for (int t = 0; t < SEQL; ++t) {
    // ---- phase 1: build bf16 feature tile A[16][1440] in LDS ----
#pragma unroll 1
    for (int idx = tid; idx < 16 * D1; idx += 256) {  // exactly 10 iters, no divergence
      int r = idx / D1;
      int i = idx - r * D1;
      float v;
      if (i < NIN) {
        const float* xp = &x[((size_t)(b0 + r) * SEQL + t) * NIN + i];
        v = *xp;
        if (t + 1 < SEQL) __builtin_prefetch(xp + NIN, 0, 0); // next step's x_t
      } else {
        v = hbuf[r * NHID + (i - NIN)];
      }
      float Bv[8];
      bspline8(v, Bv);
      __bf16* ap = &Abuf[r * ASTRIDE + i * NB];
      ap[0] = (__bf16)silu_f(v);
#pragma unroll
      for (int k = 0; k < 8; ++k) ap[1 + k] = (__bf16)Bv[k];
    }
    __syncthreads();

    // ---- phase 2: [16,1440] x [1440,16] per wave, two WMMA chains ----
    v8f acc0 = {0, 0, 0, 0, 0, 0, 0, 0};
    v8f acc1 = {0, 0, 0, 0, 0, 0, 0, 0};
#pragma unroll 1
    for (int kt = 0; kt < KT1 - 1; kt += 2) {
      v16bf a0 = loadA(Abuf, rowoff, koffA, kt);
      v16bf b0 = *(const v16bf*)(Wn + (size_t)kt * 512 + lane * 16);
      acc0 = __builtin_amdgcn_wmma_f32_16x16x32_bf16(false, a0, false, b0,
                                                     (short)0, acc0, false, false);
      v16bf a1 = loadA(Abuf, rowoff, koffA, kt + 1);
      v16bf b1 = *(const v16bf*)(Wn + (size_t)(kt + 1) * 512 + lane * 16);
      acc1 = __builtin_amdgcn_wmma_f32_16x16x32_bf16(false, a1, false, b1,
                                                     (short)0, acc1, false, false);
    }
    { // tail kt = 44
      v16bf a0 = loadA(Abuf, rowoff, koffA, KT1 - 1);
      v16bf b0 = *(const v16bf*)(Wn + (size_t)(KT1 - 1) * 512 + lane * 16);
      acc0 = __builtin_amdgcn_wmma_f32_16x16x32_bf16(false, a0, false, b0,
                                                     (short)0, acc0, false, false);
    }
    // ---- write h_new (C layout: VGPR j = row j + 8*half, col = lane&15) ----
    {
      int mb = half * 8;
      int n  = wv * 16 + (lane & 15);
#pragma unroll
      for (int j = 0; j < 8; ++j) hbuf[(mb + j) * NHID + n] = acc0[j] + acc1[j];
    }
    __syncthreads();
  }

  // ---- layer 2 (once, on final h): features then [16,1152] x [1152,16] ----
#pragma unroll 1
  for (int idx = tid; idx < 16 * D2; idx += 256) { // exactly 8 iters
    int r = idx >> 7;
    int i = idx & 127;
    float v = hbuf[r * NHID + i];
    float Bv[8];
    bspline8(v, Bv);
    __bf16* ap = &Abuf[r * ASTRIDE + i * NB];
    ap[0] = (__bf16)silu_f(v);
#pragma unroll
    for (int k = 0; k < 8; ++k) ap[1 + k] = (__bf16)Bv[k];
  }
  __syncthreads();

  if (wv == 0) { // wave-uniform branch: EXEC stays all-ones inside
    v8f acc0 = {0, 0, 0, 0, 0, 0, 0, 0};
    v8f acc1 = {0, 0, 0, 0, 0, 0, 0, 0};
#pragma unroll 1
    for (int kt = 0; kt < KT2; kt += 2) { // 36 tiles, even
      v16bf a0 = loadA(Abuf, rowoff, koffA, kt);
      v16bf b0 = *(const v16bf*)(W2p + (size_t)kt * 512 + lane * 16);
      acc0 = __builtin_amdgcn_wmma_f32_16x16x32_bf16(false, a0, false, b0,
                                                     (short)0, acc0, false, false);
      v16bf a1 = loadA(Abuf, rowoff, koffA, kt + 1);
      v16bf b1 = *(const v16bf*)(W2p + (size_t)(kt + 1) * 512 + lane * 16);
      acc1 = __builtin_amdgcn_wmma_f32_16x16x32_bf16(false, a1, false, b1,
                                                     (short)0, acc1, false, false);
    }
    int mb = half * 8;
    int n  = lane & 15;
#pragma unroll
    for (int j = 0; j < 8; ++j)
      out[(size_t)(b0 + mb + j) * NOUT + n] = acc0[j] + acc1[j];
  }
}

extern "C" void kernel_launch(void* const* d_in, const int* in_sizes, int n_in,
                              void* d_out, int out_size, void* d_ws, size_t ws_size,
                              hipStream_t stream) {
  (void)in_sizes; (void)n_in; (void)out_size; (void)ws_size;
  const float* x     = (const float*)d_in[0];
  const float* coef1 = (const float*)d_in[1];
  const float* sb1   = (const float*)d_in[2];
  const float* sp1   = (const float*)d_in[3];
  const float* coef2 = (const float*)d_in[4];
  const float* sb2   = (const float*)d_in[5];
  const float* sp2   = (const float*)d_in[6];

  __bf16* W1p = (__bf16*)d_ws;
  __bf16* W2p = (__bf16*)((char*)d_ws + W1P_BYTES);

  pack_w1<<<(W1P_ELEMS + 255) / 256, 256, 0, stream>>>(coef1, sb1, sp1, W1p);
  pack_w2<<<(W2P_ELEMS + 255) / 256, 256, 0, stream>>>(coef2, sb2, sp2, W2p);
  kan_rnn_kernel<<<BATCH / 16, 256, 0, stream>>>(x, W1p, W2p, (float*)d_out);
}
